// Tokenizer_59579786330688
// MI455X (gfx1250) — matrix operand order repaired and checked by reading, not compiled, not run on hardware
//
#include <hip/hip_runtime.h>

#define BATCH      16
#define P_PTS      16384
#define NUM_TOKENS 256
#define HIDDEN     128
#define NDIM       384
#define KNN_K      491     // int(0.03 * 16384)
#define KSUB       123     // ceil(491 / 4)

typedef _Float16 v16h __attribute__((ext_vector_type(16)));
typedef _Float16 v8h  __attribute__((ext_vector_type(8)));
typedef float    v8f  __attribute__((ext_vector_type(8)));

// ---------------------------------------------------------------------------
// Kernel 1: farthest point sampling. One block per batch; dists in LDS.
// Records `last` (scan semantics: first recorded index is 0) and ctr_pos.
// ---------------------------------------------------------------------------
__global__ __launch_bounds__(1024) void fps_kernel(const float* __restrict__ inp,
                                                   float* __restrict__ ctr_out) {
    extern __shared__ float dists[];              // 16384 floats
    __shared__ unsigned long long warpRed[32];
    __shared__ unsigned int bcastIdx;
    const int b   = blockIdx.x;
    const int tid = threadIdx.x;
    const float* binp = inp + (size_t)b * P_PTS * 6;

    for (int j = 0; j < 16; ++j) dists[j * 1024 + tid] = 1e10f;
    unsigned int last = 0;
    __syncthreads();

    for (int it = 0; it < NUM_TOKENS; ++it) {
        const float lx = binp[(size_t)last * 6 + 0];
        const float ly = binp[(size_t)last * 6 + 1];
        const float lz = binp[(size_t)last * 6 + 2];
        if (tid == 0) {
            float* c = ctr_out + ((size_t)b * NUM_TOKENS + it) * 3;
            c[0] = lx; c[1] = ly; c[2] = lz;
        }
        unsigned long long key = 0ull;            // argmax key, smaller-idx tiebreak
        for (int j = 0; j < 16; ++j) {
            const int i = j * 1024 + tid;
            const float dx = binp[(size_t)i * 6 + 0] - lx;
            const float dy = binp[(size_t)i * 6 + 1] - ly;
            const float dz = binp[(size_t)i * 6 + 2] - lz;
            float d = dx * dx + dy * dy + dz * dz;
            d = fminf(dists[i], d);
            dists[i] = d;
            const unsigned long long k =
                ((unsigned long long)__float_as_uint(d) << 32) |
                (unsigned int)(0xFFFFFFFFu - (unsigned int)i);
            key = (k > key) ? k : key;
        }
        for (int m = 16; m >= 1; m >>= 1) {       // wave32 max-reduce
            const unsigned long long o = __shfl_xor(key, m, 32);
            key = (o > key) ? o : key;
        }
        if ((tid & 31) == 0) warpRed[tid >> 5] = key;
        __syncthreads();
        if (tid < 32) {
            unsigned long long k2 = warpRed[tid];
            for (int m = 16; m >= 1; m >>= 1) {
                const unsigned long long o = __shfl_xor(k2, m, 32);
                k2 = (o > k2) ? o : k2;
            }
            if (tid == 0) bcastIdx = 0xFFFFFFFFu - (unsigned int)(k2 & 0xFFFFFFFFull);
        }
        __syncthreads();
        last = bcastIdx;
        __syncthreads();
    }
}

// ---------------------------------------------------------------------------
// Kernel 2: ordered top-491 nearest neighbours per (b,t), subsampled [::4].
// One block per row; 64 KB LDS distance array; register-cached per-thread min.
// ---------------------------------------------------------------------------
__global__ __launch_bounds__(256) void knn_kernel(const float* __restrict__ inp,
                                                  const float* __restrict__ ctr,
                                                  int* __restrict__ nn_out) {
    extern __shared__ float d2s[];                // 16384 floats
    __shared__ unsigned long long warpRed[8];
    __shared__ unsigned long long bcastKey;
    const int row = blockIdx.x;                   // b*256 + t
    const int b   = row >> 8;
    const int tid = threadIdx.x;
    const float* binp = inp + (size_t)b * P_PTS * 6;
    const float* q = ctr + (size_t)row * 3;
    const float qx = q[0], qy = q[1], qz = q[2];

    unsigned long long key = 0xFFFFFFFFFFFFFFFFull;   // argmin key
    for (int j = 0; j < 64; ++j) {
        const int i = j * 256 + tid;
        const float dx = binp[(size_t)i * 6 + 0] - qx;
        const float dy = binp[(size_t)i * 6 + 1] - qy;
        const float dz = binp[(size_t)i * 6 + 2] - qz;
        const float d = dx * dx + dy * dy + dz * dz;
        d2s[i] = d;
        const unsigned long long k =
            ((unsigned long long)__float_as_uint(d) << 32) | (unsigned int)i;
        key = (k < key) ? k : key;
    }
    __syncthreads();

    int* out = nn_out + (size_t)row * KSUB;
    for (int sel = 0; sel < KNN_K; ++sel) {
        unsigned long long k = key;
        for (int m = 16; m >= 1; m >>= 1) {
            const unsigned long long o = __shfl_xor(k, m, 32);
            k = (o < k) ? o : k;
        }
        if ((tid & 31) == 0) warpRed[tid >> 5] = k;
        __syncthreads();
        if (tid == 0) {
            unsigned long long best = warpRed[0];
            for (int wv = 1; wv < 8; ++wv) best = (warpRed[wv] < best) ? warpRed[wv] : best;
            bcastKey = best;
        }
        __syncthreads();
        const unsigned int widx = (unsigned int)(bcastKey & 0xFFFFFFFFull);
        if (tid == 0 && (sel & 3) == 0) out[sel >> 2] = (int)widx;
        if ((widx & 255u) == (unsigned)tid) {     // owner removes + rescans its slots
            d2s[widx] = __uint_as_float(0x7F800000u);
            unsigned long long nk = 0xFFFFFFFFFFFFFFFFull;
            for (int j = 0; j < 64; ++j) {
                const int i = j * 256 + tid;
                const unsigned long long kk =
                    ((unsigned long long)__float_as_uint(d2s[i]) << 32) | (unsigned int)i;
                nk = (kk < nk) ? kk : nk;
            }
            key = nk;
        }
        __syncthreads();
    }
}

// ---------------------------------------------------------------------------
// Closed-form symmetric 3x3 eigen solve (min & max eigenvectors).
// A packed: a00 a01 a02 a11 a12 a22
// ---------------------------------------------------------------------------
__device__ inline void eigvec3(const float* A, float lam, float* v) {
    const float m00 = A[0] - lam, m01 = A[1], m02 = A[2];
    const float m11 = A[3] - lam, m12 = A[4], m22 = A[5] - lam;
    float c0x = m01 * m12 - m02 * m11, c0y = m02 * m01 - m00 * m12, c0z = m00 * m11 - m01 * m01;
    float c1x = m01 * m22 - m02 * m12, c1y = m02 * m02 - m00 * m22, c1z = m00 * m12 - m01 * m02;
    float c2x = m11 * m22 - m12 * m12, c2y = m02 * m12 - m01 * m22, c2z = m01 * m12 - m11 * m02;
    const float n0 = c0x * c0x + c0y * c0y + c0z * c0z;
    const float n1 = c1x * c1x + c1y * c1y + c1z * c1z;
    const float n2 = c2x * c2x + c2y * c2y + c2z * c2z;
    float bx = c0x, by = c0y, bz = c0z, bn = n0;
    if (n1 > bn) { bx = c1x; by = c1y; bz = c1z; bn = n1; }
    if (n2 > bn) { bx = c2x; by = c2y; bz = c2z; bn = n2; }
    if (bn < 1e-30f) { v[0] = 0.0f; v[1] = 0.0f; v[2] = 1.0f; return; }
    const float inv = rsqrtf(bn);
    v[0] = bx * inv; v[1] = by * inv; v[2] = bz * inv;
}

__device__ inline void eig3_minmax(const float* A, float* nvec, float* zvec) {
    const float a00 = A[0], a01 = A[1], a02 = A[2], a11 = A[3], a12 = A[4], a22 = A[5];
    const float p1 = a01 * a01 + a02 * a02 + a12 * a12;
    const float q = (a00 + a11 + a22) * (1.0f / 3.0f);
    float lmin, lmax;
    if (p1 < 1e-18f) {
        lmin = fminf(a00, fminf(a11, a22));
        lmax = fmaxf(a00, fmaxf(a11, a22));
    } else {
        const float b00 = a00 - q, b11 = a11 - q, b22 = a22 - q;
        const float p2 = b00 * b00 + b11 * b11 + b22 * b22 + 2.0f * p1;
        const float p = sqrtf(p2 * (1.0f / 6.0f));
        const float ip = 1.0f / p;
        const float c00 = b00 * ip, c01 = a01 * ip, c02 = a02 * ip;
        const float c11 = b11 * ip, c12 = a12 * ip, c22 = b22 * ip;
        const float detB = c00 * (c11 * c22 - c12 * c12)
                         - c01 * (c01 * c22 - c12 * c02)
                         + c02 * (c01 * c12 - c11 * c02);
        float r = 0.5f * detB;
        r = fminf(1.0f, fmaxf(-1.0f, r));
        const float phi = acosf(r) * (1.0f / 3.0f);
        lmax = q + 2.0f * p * __cosf(phi);
        lmin = q + 2.0f * p * __cosf(phi + 2.0943951023931953f); // + 2*pi/3
    }
    eigvec3(A, lmin, nvec);
    eigvec3(A, lmax, zvec);
}

// ---------------------------------------------------------------------------
// Kernel 3 (fused): gather neighbours, weights, covariance, eigh, LRF,
// rotate to tps, MLP1 (6->128, K=6 is too small for WMMA) + max-pool.
// Writes lrfs (f32) and g (f16 for the WMMA GEMM).
// ---------------------------------------------------------------------------
__global__ __launch_bounds__(128) void token_geom_mlp1_kernel(
    const float* __restrict__ inp, const float* __restrict__ ctr,
    const int* __restrict__ nn, const float* __restrict__ W1,
    const float* __restrict__ b1, float* __restrict__ lrf_out,
    _Float16* __restrict__ g16) {
    __shared__ float sTps[128][6];
    __shared__ float sRed[128][6];
    __shared__ float sR[9];
    const int row = blockIdx.x;
    const int b   = row >> 8;
    const int t   = threadIdx.x;
    const bool valid = (t < KSUB);
    const float* binp = inp + (size_t)b * P_PTS * 6;
    const float* c = ctr + (size_t)row * 3;

    float px = 0.f, py = 0.f, pz = 0.f, ox = 0.f, oy = 0.f, oz = 0.f, nrm = -1e30f;
    if (valid) {
        const int i = nn[(size_t)row * KSUB + t];
        const float* p = binp + (size_t)i * 6;
        px = p[0] - c[0]; py = p[1] - c[1]; pz = p[2] - c[2];
        ox = p[3]; oy = p[4]; oz = p[5];
        nrm = sqrtf(px * px + py * py + pz * pz);
    }
    // max norm
    sRed[t][0] = nrm;
    __syncthreads();
    for (int s = 64; s > 0; s >>= 1) {
        if (t < s) sRed[t][0] = fmaxf(sRed[t][0], sRed[t + s][0]);
        __syncthreads();
    }
    const float maxn = sRed[0][0];
    __syncthreads();
    // weight sum
    const float w = valid ? (maxn - nrm) : 0.0f;
    sRed[t][0] = w;
    __syncthreads();
    for (int s = 64; s > 0; s >>= 1) {
        if (t < s) sRed[t][0] += sRed[t + s][0];
        __syncthreads();
    }
    const float wn = w / sRed[0][0];
    __syncthreads();
    // weighted covariance of 100*lps_pos
    const float sx = 100.f * px, sy = 100.f * py, sz = 100.f * pz;
    sRed[t][0] = wn * sx * sx; sRed[t][1] = wn * sx * sy; sRed[t][2] = wn * sx * sz;
    sRed[t][3] = wn * sy * sy; sRed[t][4] = wn * sy * sz; sRed[t][5] = wn * sz * sz;
    __syncthreads();
    for (int s = 64; s > 0; s >>= 1) {
        if (t < s) for (int j = 0; j < 6; ++j) sRed[t][j] += sRed[t + s][j];
        __syncthreads();
    }
    if (t == 0) {
        float A[6] = {sRed[0][0], sRed[0][1], sRed[0][2], sRed[0][3], sRed[0][4], sRed[0][5]};
        float n0[3], z0[3];
        eig3_minmax(A, n0, z0);
        sR[0] = n0[0]; sR[1] = n0[1]; sR[2] = n0[2];
        sR[3] = z0[0]; sR[4] = z0[1]; sR[5] = z0[2];
    }
    __syncthreads();
    const float nx = sR[0], ny = sR[1], nz = sR[2];
    const float zx = sR[3], zy = sR[4], zz = sR[5];
    __syncthreads();
    // disambiguation counts
    sRed[t][0] = (valid && (nx * px + ny * py + nz * pz) > 0.f) ? 1.f : 0.f;
    sRed[t][1] = (valid && (zx * px + zy * py + zz * pz) > 0.f) ? 1.f : 0.f;
    __syncthreads();
    for (int s = 64; s > 0; s >>= 1) {
        if (t < s) { sRed[t][0] += sRed[t + s][0]; sRed[t][1] += sRed[t + s][1]; }
        __syncthreads();
    }
    if (t == 0) {
        float fnx = nx, fny = ny, fnz = nz, fzx = zx, fzy = zy, fzz = zz;
        if (2.0f * sRed[0][0] < (float)KSUB) { fnx = -fnx; fny = -fny; fnz = -fnz; }
        if (2.0f * sRed[0][1] < (float)KSUB) { fzx = -fzx; fzy = -fzy; fzz = -fzz; }
        const float yx = fny * fzz - fnz * fzy;
        const float yy = fnz * fzx - fnx * fzz;
        const float yz = fnx * fzy - fny * fzx;
        sR[0] = fnx; sR[1] = yx; sR[2] = fzx;     // R[c*3+d], columns n,y,z
        sR[3] = fny; sR[4] = yy; sR[5] = fzy;
        sR[6] = fnz; sR[7] = yz; sR[8] = fzz;
        float* lo = lrf_out + (size_t)row * 9;
        for (int j = 0; j < 9; ++j) lo[j] = sR[j];
    }
    __syncthreads();
    if (valid) {
        for (int d = 0; d < 3; ++d) {
            sTps[t][d]     = px * sR[0 * 3 + d] + py * sR[1 * 3 + d] + pz * sR[2 * 3 + d];
            sTps[t][3 + d] = ox * sR[0 * 3 + d] + oy * sR[1 * 3 + d] + oz * sR[2 * 3 + d];
        }
    }
    __syncthreads();
    // MLP1 column per thread + max-pool over k (relu commutes with max)
    const int h = t;
    float w1c[6];
    for (int cd = 0; cd < 6; ++cd) w1c[cd] = W1[cd * HIDDEN + h];
    const float bb = b1[h];
    float m = -1e30f;
    for (int k = 0; k < KSUB; ++k) {
        float acc = bb;
        for (int cd = 0; cd < 6; ++cd) acc = fmaf(sTps[k][cd], w1c[cd], acc);
        m = fmaxf(m, acc);
    }
    m = fmaxf(m, 0.0f);
    g16[(size_t)row * HIDDEN + h] = (_Float16)m;
}

// ---------------------------------------------------------------------------
// Kernel 4: transpose + convert W2 (128x384 f32) -> W2T (384x128 f16)
// so WMMA B fragments become contiguous 32-byte loads.
// ---------------------------------------------------------------------------
__global__ void w2t_kernel(const float* __restrict__ W2, _Float16* __restrict__ w2t) {
    const int idx = blockIdx.x * 256 + threadIdx.x;
    if (idx < HIDDEN * NDIM) {
        const int n = idx / HIDDEN, k = idx % HIDDEN;
        w2t[idx] = (_Float16)W2[(size_t)k * NDIM + n];
    }
}

// ---------------------------------------------------------------------------
// Kernel 5: MLP2 GEMM  tokens = g(4096x128) @ W2(128x384) + b2
// v_wmma_f32_16x16x32_f16 : one 16x16 tile per wave, 4 K-steps of 32.
// A as loaded here (contiguous k, same as B): lane<16 -> K 0-15; lane>=16 -> K 16-31.
// B as loaded here (contiguous k, same as A): lane<16 -> col=lane, K 0-15; lane>=16 -> K 16-31.
// ---------------------------------------------------------------------------
__global__ __launch_bounds__(256) void mlp2_wmma_kernel(
    const _Float16* __restrict__ g16, const _Float16* __restrict__ w2t,
    const float* __restrict__ b2, float* __restrict__ tok) {
    const int tid     = threadIdx.x;
    const int wave    = tid >> 5;
    const int lane    = tid & 31;
    const int tileId  = blockIdx.x * 8 + wave;
    const int mTile   = tileId & 255;             // 4096/16 = 256 m-tiles
    const int nTile   = tileId >> 8;              // 384/16  = 24 n-tiles
    const int laneLow = lane & 15;
    const int hiHalf  = lane >> 4;

    const int rowA = mTile * 16 + laneLow;
    const int colB = nTile * 16 + laneLow;

    v8f acc = {};
    union { v16h v; v8h h[2]; } a;
#pragma unroll
    for (int kk = 0; kk < HIDDEN; kk += 32) {
        const int kA = kk + hiHalf * 16;
        a.h[0] = *(const v8h*)(g16 + (size_t)rowA * HIDDEN + kA);
        a.h[1] = *(const v8h*)(g16 + (size_t)rowA * HIDDEN + kA + 8);
        const int kB = kk + hiHalf * 16;
        const v16h bfrag = *(const v16h*)(w2t + (size_t)colB * HIDDEN + kB);
        acc = __builtin_amdgcn_wmma_f32_16x16x32_f16(
            false, a.v, false, bfrag, (short)0, acc, false, false);
    }
    const float bias = b2[colB];
#pragma unroll
    for (int r = 0; r < 8; ++r) {
        const int m = mTile * 16 + r + hiHalf * 8;
        tok[(size_t)m * NDIM + colB] = acc[r] + bias;
    }
}

// ---------------------------------------------------------------------------
extern "C" void kernel_launch(void* const* d_in, const int* in_sizes, int n_in,
                              void* d_out, int out_size, void* d_ws, size_t ws_size,
                              hipStream_t stream) {
    (void)in_sizes; (void)n_in; (void)out_size; (void)ws_size;
    const float* inp = (const float*)d_in[0];
    const float* W1  = (const float*)d_in[1];
    const float* b1  = (const float*)d_in[2];
    const float* W2  = (const float*)d_in[3];
    const float* b2  = (const float*)d_in[4];

    float* tok = (float*)d_out;                                   // 16*256*384
    float* ctr = tok + (size_t)BATCH * NUM_TOKENS * NDIM;         // 16*256*3
    float* lrf = ctr + (size_t)BATCH * NUM_TOKENS * 3;            // 16*256*9

    char* ws = (char*)d_ws;
    int*      nn  = (int*)ws;                                     // 4096*123*4 < 2 MB
    _Float16* g16 = (_Float16*)(ws + (2u << 20));                 // 4096*128*2 = 1 MB
    _Float16* w2t = (_Float16*)(ws + (3u << 20));                 // 384*128*2

    fps_kernel<<<BATCH, 1024, P_PTS * sizeof(float), stream>>>(inp, ctr);
    knn_kernel<<<BATCH * NUM_TOKENS, 256, P_PTS * sizeof(float), stream>>>(inp, ctr, nn);
    token_geom_mlp1_kernel<<<BATCH * NUM_TOKENS, 128, 0, stream>>>(inp, ctr, nn, W1, b1, lrf, g16);
    w2t_kernel<<<(HIDDEN * NDIM + 255) / 256, 256, 0, stream>>>(W2, w2t);
    mlp2_wmma_kernel<<<(BATCH * NUM_TOKENS / 16) * (NDIM / 16) / 8, 256, 0, stream>>>(
        g16, w2t, b2, tok);
}
